// SpanAsteModel_47742856462634
// MI455X (gfx1250) — matrix operand order repaired and checked by reading, not compiled, not run on hardware
//
#include <hip/hip_runtime.h>
#include <hip/hip_bf16.h>

typedef __attribute__((ext_vector_type(16))) __bf16 v16bf;
typedef __attribute__((ext_vector_type(8)))  __bf16 v8bf;
typedef __attribute__((ext_vector_type(8)))  float  v8f;

#define NS    996      // number of spans per batch (S=128, max_len=8)
#define NB    8        // batch
#define SEQ   128
#define HDIM  768
#define NZ    64       // pruned spans per class
#define HID   150
#define HIDP  160      // HID padded to 16/32
#define WDIM  20
#define SDIM  1556     // span rep dim
#define SKP   1568     // SDIM padded to 32
#define DDIM  128
#define NBINS 10
#define NEG_INF (-3.402823466e38f)

// ---------------- span index tables -----------------------------------------
__global__ void k_span_idx(int* starts, int* ends, int* widths) {
    int n = blockIdx.x * blockDim.x + threadIdx.x;
    if (n >= NS) return;
    int off = 0;
    for (int w = 1; w <= 8; ++w) {
        int cnt = SEQ - w + 1;
        if (n < off + cnt) { int s = n - off; starts[n] = s; ends[n] = s + w - 1; widths[n] = w - 1; return; }
        off += cnt;
    }
}

// ---------------- weight convert: W[k0+k][n] (f32,ld=ldw) -> Wt[n][k] bf16 ---
__global__ void k_wt(const float* __restrict__ W, int ldw, int k0, int K, int N,
                     __bf16* __restrict__ Wt, int Kp, int Np) {
    long i = (long)blockIdx.x * blockDim.x + threadIdx.x;
    long tot = (long)Np * Kp;
    if (i >= tot) return;
    int n = (int)(i / Kp), k = (int)(i % Kp);
    float v = (n < N && k < K) ? W[(long)(k0 + k) * ldw + n] : 0.f;
    Wt[i] = (__bf16)v;
}

// ---------------- build span representation rows (bf16, K-padded) -----------
__global__ void k_build_spans(const float* __restrict__ x, const float* __restrict__ wemb,
                              const int* __restrict__ starts, const int* __restrict__ ends,
                              const int* __restrict__ widths, __bf16* __restrict__ spans) {
    int row = blockIdx.x;                 // 0 .. NB*NS-1
    int b = row / NS, n = row % NS;
    const float* xb = x + (long)b * SEQ * HDIM;
    __bf16* out = spans + (long)row * SKP;
    int s = starts[n], e = ends[n], w = widths[n];
    for (int j = threadIdx.x; j < SKP; j += blockDim.x) {
        float v;
        if (j < HDIM)            v = xb[(long)s * HDIM + j];
        else if (j < 2 * HDIM)   v = xb[(long)e * HDIM + (j - HDIM)];
        else if (j < SDIM)       v = wemb[w * WDIM + (j - 2 * HDIM)];
        else                     v = 0.f;
        out[j] = (__bf16)v;
    }
}

// ---------------- fragment helpers ------------------------------------------
// A fragment (16-bit A layout): lane-half kh reads runs [8kh,8kh+8) and [16+8kh,+8)
__device__ __forceinline__ v16bf load_afrag(const __bf16* p) {
    v8bf lo = *(const v8bf*)p;
    v8bf hi = *(const v8bf*)(p + 16);
    v16bf a;
#pragma unroll
    for (int i = 0; i < 8; ++i) { a[i] = lo[i]; a[8 + i] = hi[i]; }
    return a;
}

// ---------------- WMMA GEMM, 2x2 tiles (32x32 output) per wave ---------------
// A: bf16 [M,Kp] row-major (optional row gather); Wt: bf16 [Np,Kp] col-of-W contiguous
__global__ __launch_bounds__(128)
void k_gemm22(const __bf16* __restrict__ A, const int* __restrict__ gather, int Kp,
              const __bf16* __restrict__ Wt, int Np,
              const float* __restrict__ bias, int nvalid, int relu,
              float* __restrict__ Cf, __bf16* __restrict__ Cb, int ldc, int Mt32) {
    int wave = blockIdx.x * (blockDim.x >> 5) + (threadIdx.x >> 5);
    int Nt32 = Np >> 5;
    if (wave >= Mt32 * Nt32) return;          // whole wave exits together
    int tm = wave / Nt32, tn = wave % Nt32;
    int lane = threadIdx.x & 31;
    int lm = lane & 15, kh = lane >> 4;

    int ar0 = tm * 32 + lm, ar1 = ar0 + 16;
    long ap0 = gather ? (long)gather[ar0] : (long)ar0;
    long ap1 = gather ? (long)gather[ar1] : (long)ar1;
    const __bf16* A0 = A + ap0 * Kp + 8 * kh;
    const __bf16* A1 = A + ap1 * Kp + 8 * kh;
    // B fragment: same layout as A, runs [8kh,8kh+8) and [16+8kh,+8)
    const __bf16* B0 = Wt + (long)(tn * 32 + lm) * Kp + 8 * kh;
    const __bf16* B1 = Wt + (long)(tn * 32 + 16 + lm) * Kp + 8 * kh;

    v8f c00 = {}, c01 = {}, c10 = {}, c11 = {};
    for (int kb = 0; kb < Kp; kb += 32) {
        if (kb + 256 < Kp) {                  // look-ahead prefetch (global_prefetch_b8)
            __builtin_prefetch(A0 + kb + 256, 0, 1);
            __builtin_prefetch(A1 + kb + 256, 0, 1);
        }
        v16bf a0 = load_afrag(A0 + kb);
        v16bf a1 = load_afrag(A1 + kb);
        v16bf b0 = load_afrag(B0 + kb);
        v16bf b1 = load_afrag(B1 + kb);
        c00 = __builtin_amdgcn_wmma_f32_16x16x32_bf16(false, a0, false, b0, (short)0, c00, false, false);
        c01 = __builtin_amdgcn_wmma_f32_16x16x32_bf16(false, a0, false, b1, (short)0, c01, false, false);
        c10 = __builtin_amdgcn_wmma_f32_16x16x32_bf16(false, a1, false, b0, (short)0, c10, false, false);
        c11 = __builtin_amdgcn_wmma_f32_16x16x32_bf16(false, a1, false, b1, (short)0, c11, false, false);
    }

    auto store = [&](v8f c, int ci, int cj) {
        int col = tn * 32 + cj * 16 + lm;
        float bv = (bias && col < nvalid) ? bias[col] : 0.f;
#pragma unroll
        for (int r = 0; r < 8; ++r) {
            int m = tm * 32 + ci * 16 + r + 8 * kh;   // C layout: lanes16-31 hold M=8..15
            float v = c[r] + bv;
            if (relu) v = v > 0.f ? v : 0.f;
            if (Cf) Cf[(long)m * ldc + col] = v;
            else    Cb[(long)m * ldc + col] = (__bf16)v;
        }
    };
    store(c00, 0, 0); store(c01, 0, 1); store(c10, 1, 0); store(c11, 1, 1);
}

// ---------------- WMMA GEMM, 1x1 tile per wave (for Np=16 logit layers) ------
__global__ __launch_bounds__(128)
void k_gemm(const __bf16* __restrict__ A, const int* __restrict__ gather, int Kp,
            const __bf16* __restrict__ Wt, int Np,
            const float* __restrict__ bias, int nvalid, int relu,
            float* __restrict__ Cf, __bf16* __restrict__ Cb, int ldc, int Mtiles) {
    int wave = blockIdx.x * (blockDim.x >> 5) + (threadIdx.x >> 5);
    int Ntiles = Np >> 4;
    if (wave >= Mtiles * Ntiles) return;
    int tm = wave / Ntiles, tn = wave % Ntiles;
    int lane = threadIdx.x & 31;
    int lm = lane & 15, kh = lane >> 4;

    int arow = tm * 16 + lm;
    long aphys = gather ? (long)gather[arow] : (long)arow;
    const __bf16* Ar = A + aphys * Kp + 8 * kh;
    const __bf16* Br = Wt + (long)(tn * 16 + lm) * Kp + 8 * kh;

    v8f c = {};
    for (int kb = 0; kb < Kp; kb += 32) {
        v16bf a = load_afrag(Ar + kb);
        v16bf b = load_afrag(Br + kb);
        c = __builtin_amdgcn_wmma_f32_16x16x32_bf16(false, a, false, b, (short)0, c, false, false);
    }
    int col = tn * 16 + lm;
    float bv = (bias && col < nvalid) ? bias[col] : 0.f;
#pragma unroll
    for (int r = 0; r < 8; ++r) {
        int m = tm * 16 + r + 8 * kh;
        float v = c[r] + bv;
        if (relu) v = v > 0.f ? v : 0.f;
        if (Cf) Cf[(long)m * ldc + col] = v;
        else    Cb[(long)m * ldc + col] = (__bf16)v;
    }
}

// ---------------- small-row softmax -----------------------------------------
__global__ void k_softmax(const float* __restrict__ logits, int ld, int C, long rows,
                          float* __restrict__ out) {
    long r = (long)blockIdx.x * blockDim.x + threadIdx.x;
    if (r >= rows) return;
    const float* l = logits + r * ld;
    float mx = l[0];
    for (int c = 1; c < C; ++c) mx = fmaxf(mx, l[c]);
    float e[4], s = 0.f;
    for (int c = 0; c < C; ++c) { e[c] = __expf(l[c] - mx); s += e[c]; }
    float inv = 1.f / s;
    for (int c = 0; c < C; ++c) out[r * C + c] = e[c] * inv;
}

// ---------------- top-64 per (batch,class): iterative argmax ----------------
__global__ __launch_bounds__(256)
void k_topk(const float* __restrict__ sprob, int* __restrict__ tIdx, int* __restrict__ oIdx,
            int* __restrict__ tRow, int* __restrict__ oRow) {
    __shared__ float vals[NS];
    __shared__ float rv[256];
    __shared__ int   ri[256];
    int b = blockIdx.x >> 1;
    int cls = (blockIdx.x & 1) ? 2 : 1;                  // 1 = ASPECT, 2 = OPINION
    int tid = threadIdx.x;
    for (int n = tid; n < NS; n += 256)
        vals[n] = sprob[((long)b * NS + n) * 3 + cls];
    __syncthreads();
    int* outIdx = (blockIdx.x & 1) ? oIdx : tIdx;
    int* outRow = (blockIdx.x & 1) ? oRow : tRow;
    for (int it = 0; it < NZ; ++it) {
        float bv = NEG_INF; int bi = NS;
        for (int n = tid; n < NS; n += 256) {
            float v = vals[n];
            if (v > bv || (v == bv && n < bi)) { bv = v; bi = n; }
        }
        rv[tid] = bv; ri[tid] = bi;
        __syncthreads();
        for (int s = 128; s > 0; s >>= 1) {
            if (tid < s) {
                float v2 = rv[tid + s]; int i2 = ri[tid + s];
                if (v2 > rv[tid] || (v2 == rv[tid] && i2 < ri[tid])) { rv[tid] = v2; ri[tid] = i2; }
            }
            __syncthreads();
        }
        if (tid == 0) {
            int idx = ri[0];
            outIdx[b * NZ + it] = idx;
            outRow[b * NZ + it] = b * NS + idx;
            vals[idx] = NEG_INF;
        }
        __syncthreads();
    }
}

// ---------------- D1[bucket] = dist_emb[bucket] @ pW1[3112:3240] ------------
__global__ void k_d1(const float* __restrict__ dist_emb, const float* __restrict__ pW1,
                     float* __restrict__ D1) {
    int i = blockIdx.x * blockDim.x + threadIdx.x;
    if (i >= NBINS * HIDP) return;
    int r = i / HIDP, n = i % HIDP;
    float acc = 0.f;
    if (n < HID)
        for (int k = 0; k < DDIM; ++k)
            acc += dist_emb[r * DDIM + k] * pW1[(long)(2 * SDIM + k) * HID + n];
    D1[i] = acc;
}

// ---------------- pair layer-1 recombination: relu(T1+O1+D1+pb1) -> bf16 ----
__global__ void k_combine(const float* __restrict__ T1, const float* __restrict__ O1,
                          const float* __restrict__ D1, const float* __restrict__ pb1,
                          const int* __restrict__ tIdx, const int* __restrict__ oIdx,
                          const int* __restrict__ starts, const int* __restrict__ ends,
                          __bf16* __restrict__ h1p) {
    long i = (long)blockIdx.x * blockDim.x + threadIdx.x;
    if (i >= (long)NB * NZ * NZ * HIDP) return;
    int n = (int)(i % HIDP);
    long p = i / HIDP;
    int oj = (int)(p % NZ);
    int ti = (int)((p / NZ) % NZ);
    int b  = (int)(p / (NZ * NZ));
    int tspan = tIdx[b * NZ + ti], ospan = oIdx[b * NZ + oj];
    int aa = starts[tspan], bb = ends[tspan], cc = starts[ospan], dd = ends[ospan];
    int d1 = bb - cc; if (d1 < 0) d1 = -d1;
    int d2 = aa - dd; if (d2 < 0) d2 = -d2;
    int dist = d1 < d2 ? d1 : d2;
    const int bins[NBINS] = {0, 1, 2, 3, 4, 5, 8, 16, 32, 64};
    int bucket = 0;
#pragma unroll
    for (int k = 1; k < NBINS; ++k) if (dist >= bins[k]) bucket = k;
    float v = T1[(long)(b * NZ + ti) * HIDP + n] + O1[(long)(b * NZ + oj) * HIDP + n]
            + D1[bucket * HIDP + n] + ((n < HID) ? pb1[n] : 0.f);
    h1p[i] = (__bf16)(v > 0.f ? v : 0.f);
}

// ---------------- indices -> float tail of d_out ----------------------------
__global__ void k_idx2f(const int* __restrict__ tIdx, const int* __restrict__ oIdx,
                        float* __restrict__ out) {
    int i = blockIdx.x * blockDim.x + threadIdx.x;
    if (i >= 2 * NB * NZ) return;
    out[i] = (float)(i < NB * NZ ? tIdx[i] : oIdx[i - NB * NZ]);
}

// ============================================================================
extern "C" void kernel_launch(void* const* d_in, const int* in_sizes, int n_in,
                              void* d_out, int out_size, void* d_ws, size_t ws_size,
                              hipStream_t stream) {
    const float* x        = (const float*)d_in[0];
    const float* wemb     = (const float*)d_in[1];
    const float* sW1      = (const float*)d_in[2];
    const float* sb1      = (const float*)d_in[3];
    const float* sW2      = (const float*)d_in[4];
    const float* sb2      = (const float*)d_in[5];
    const float* sW3      = (const float*)d_in[6];
    const float* sb3      = (const float*)d_in[7];
    const float* dist_emb = (const float*)d_in[8];
    const float* pW1      = (const float*)d_in[9];
    const float* pb1      = (const float*)d_in[10];
    const float* pW2      = (const float*)d_in[11];
    const float* pb2      = (const float*)d_in[12];
    const float* pW3      = (const float*)d_in[13];
    const float* pb3      = (const float*)d_in[14];

    float* out = (float*)d_out;
    float* out_sprob = out;                                    // 8*996*3
    float* out_cprob = out + NB * NS * 3;                      // 8*4096*4
    float* out_idx   = out + NB * NS * 3 + NB * NZ * NZ * 4;   // 2*512

    // workspace bump allocator (256B aligned)
    char*  base = (char*)d_ws;
    size_t off  = 0;
    auto alloc = [&](size_t bytes) -> void* {
        void* p = base + off;
        off = (off + bytes + 255) & ~(size_t)255;
        return p;
    };
    const long Ms = (long)NB * NS;         // 7968 span rows (= 249*32)
    const long Mp = (long)NB * NZ * NZ;    // 32768 pair rows

    int*    starts = (int*)alloc(NS * 4);
    int*    ends   = (int*)alloc(NS * 4);
    int*    widths = (int*)alloc(NS * 4);
    __bf16* spans  = (__bf16*)alloc((size_t)Ms * SKP * 2);
    __bf16* sW1t   = (__bf16*)alloc((size_t)HIDP * SKP * 2);
    __bf16* sW2t   = (__bf16*)alloc((size_t)HIDP * HIDP * 2);
    __bf16* sW3t   = (__bf16*)alloc((size_t)16 * HIDP * 2);
    __bf16* pW1at  = (__bf16*)alloc((size_t)HIDP * SKP * 2);
    __bf16* pW1bt  = (__bf16*)alloc((size_t)HIDP * SKP * 2);
    __bf16* pW2t   = (__bf16*)alloc((size_t)HIDP * HIDP * 2);
    __bf16* pW3t   = (__bf16*)alloc((size_t)16 * HIDP * 2);
    __bf16* h1     = (__bf16*)alloc((size_t)Ms * HIDP * 2);
    __bf16* h2     = (__bf16*)alloc((size_t)Ms * HIDP * 2);
    float*  slog   = (float*)alloc((size_t)Ms * 16 * 4);
    int*    tIdx   = (int*)alloc(NB * NZ * 4);
    int*    oIdx   = (int*)alloc(NB * NZ * 4);
    int*    tRow   = (int*)alloc(NB * NZ * 4);
    int*    oRow   = (int*)alloc(NB * NZ * 4);
    float*  T1     = (float*)alloc((size_t)NB * NZ * HIDP * 4);
    float*  O1     = (float*)alloc((size_t)NB * NZ * HIDP * 4);
    float*  D1     = (float*)alloc((size_t)NBINS * HIDP * 4);
    __bf16* h1p    = (__bf16*)alloc((size_t)Mp * HIDP * 2);
    __bf16* h2p    = (__bf16*)alloc((size_t)Mp * HIDP * 2);
    float*  plog   = (float*)alloc((size_t)Mp * 16 * 4);
    (void)ws_size; (void)in_sizes; (void)n_in; (void)out_size;

    // 1) index tables + weight conversion (bf16, transposed, padded)
    hipLaunchKernelGGL(k_span_idx, dim3(4), dim3(256), 0, stream, starts, ends, widths);
    auto cvt = [&](const float* W, int ldw, int k0, int K, int N, __bf16* Wt, int Kp, int Np) {
        long tot = (long)Np * Kp;
        hipLaunchKernelGGL(k_wt, dim3((unsigned)((tot + 255) / 256)), dim3(256), 0, stream,
                           W, ldw, k0, K, N, Wt, Kp, Np);
    };
    cvt(sW1, HID, 0,    SDIM, HID, sW1t, SKP,  HIDP);
    cvt(sW2, HID, 0,    HID,  HID, sW2t, HIDP, HIDP);
    cvt(sW3, 3,   0,    HID,  3,   sW3t, HIDP, 16);
    cvt(pW1, HID, 0,    SDIM, HID, pW1at, SKP, HIDP);
    cvt(pW1, HID, SDIM, SDIM, HID, pW1bt, SKP, HIDP);
    cvt(pW2, HID, 0,    HID,  HID, pW2t, HIDP, HIDP);
    cvt(pW3, 4,   0,    HID,  4,   pW3t, HIDP, 16);

    // 2) span representation (bf16)
    hipLaunchKernelGGL(k_build_spans, dim3((unsigned)Ms), dim3(256), 0, stream,
                       x, wemb, starts, ends, widths, spans);

    auto gemm22 = [&](const __bf16* A, const int* gth, int Kp, const __bf16* Wt, int Np,
                      const float* bias, int nvalid, int relu,
                      float* Cf, __bf16* Cb, int ldc, int Mt32) {
        int waves = Mt32 * (Np >> 5);
        hipLaunchKernelGGL(k_gemm22, dim3((unsigned)((waves + 3) / 4)), dim3(128), 0, stream,
                           A, gth, Kp, Wt, Np, bias, nvalid, relu, Cf, Cb, ldc, Mt32);
    };
    auto gemm11 = [&](const __bf16* A, const int* gth, int Kp, const __bf16* Wt, int Np,
                      const float* bias, int nvalid, int relu,
                      float* Cf, __bf16* Cb, int ldc, int Mtiles) {
        int waves = Mtiles * (Np >> 4);
        hipLaunchKernelGGL(k_gemm, dim3((unsigned)((waves + 3) / 4)), dim3(128), 0, stream,
                           A, gth, Kp, Wt, Np, bias, nvalid, relu, Cf, Cb, ldc, Mtiles);
    };

    // 3) span FFNN
    gemm22(spans, nullptr, SKP,  sW1t, HIDP, sb1, HID, 1, nullptr, h1, HIDP, (int)(Ms / 32));
    gemm22(h1,    nullptr, HIDP, sW2t, HIDP, sb2, HID, 1, nullptr, h2, HIDP, (int)(Ms / 32));
    gemm11(h2,    nullptr, HIDP, sW3t, 16,   sb3, 3,   0, slog, nullptr, 16, (int)(Ms / 16));
    hipLaunchKernelGGL(k_softmax, dim3((unsigned)((Ms + 255) / 256)), dim3(256), 0, stream,
                       slog, 16, 3, Ms, out_sprob);

    // 4) top-64 per (batch, class)
    hipLaunchKernelGGL(k_topk, dim3(2 * NB), dim3(256), 0, stream,
                       out_sprob, tIdx, oIdx, tRow, oRow);

    // 5) factorized pair layer 1: T1, O1 (gathered span rows), D1 (10 buckets)
    gemm22(spans, tRow, SKP, pW1at, HIDP, nullptr, 0, 0, T1, nullptr, HIDP, (NB * NZ) / 32);
    gemm22(spans, oRow, SKP, pW1bt, HIDP, nullptr, 0, 0, O1, nullptr, HIDP, (NB * NZ) / 32);
    hipLaunchKernelGGL(k_d1, dim3((NBINS * HIDP + 255) / 256), dim3(256), 0, stream,
                       dist_emb, pW1, D1);
    {
        long tot = Mp * HIDP;
        hipLaunchKernelGGL(k_combine, dim3((unsigned)((tot + 255) / 256)), dim3(256), 0, stream,
                           T1, O1, D1, pb1, tIdx, oIdx, starts, ends, h1p);
    }

    // 6) pair FFNN layers 2,3 + softmax
    gemm22(h1p, nullptr, HIDP, pW2t, HIDP, pb2, HID, 1, nullptr, h2p, HIDP, (int)(Mp / 32));
    gemm11(h2p, nullptr, HIDP, pW3t, 16,   pb3, 4,   0, plog, nullptr, 16, (int)(Mp / 16));
    hipLaunchKernelGGL(k_softmax, dim3((unsigned)((Mp + 255) / 256)), dim3(256), 0, stream,
                       plog, 16, 4, Mp, out_cprob);

    // 7) indices to output tail
    hipLaunchKernelGGL(k_idx2f, dim3((2 * NB * NZ + 255) / 256), dim3(256), 0, stream,
                       tIdx, oIdx, out_idx);
}
